// GATLayer_66692252172956
// MI455X (gfx1250) — matrix operand order repaired and checked by reading, not compiled, not run on hardware
//
#include <hip/hip_runtime.h>
#include <hip/hip_bf16.h>

typedef __attribute__((ext_vector_type(16))) _Float16 v16h;
typedef __attribute__((ext_vector_type(8)))  float    v8f;

#define NEG_SLOPE 0.2f

__device__ __forceinline__ float lrelu(float v) { return v > 0.0f ? v : NEG_SLOPE * v; }

// ---------------------------------------------------------------------------
// 0) Wt[o][k] = (f16) W[k][o]  — one-time transpose so WMMA B-fragments are
//    contiguous in LDS (ds_load_b128 instead of strided ds_load_u16).
// ---------------------------------------------------------------------------
__global__ __launch_bounds__(256) void gat_w_f16t(
    const float* __restrict__ W, _Float16* __restrict__ Wt)
{
    int i = blockIdx.x * 256 + threadIdx.x;        // 16384 elements
    int k = i >> 7, o = i & 127;
    Wt[o * 128 + k] = (_Float16)W[k * 128 + o];
}

// ---------------------------------------------------------------------------
// 1) h = x @ W via v_wmma_f32_16x16x32_f16, with FUSED a_src/a_dst epilogue.
//    Block = 256 threads = 8 waves; block owns 32 rows; wave w owns cols
//    [16w,16w+16) for BOTH 16-row tiles (B fragment reused across tiles).
//    Since the block covers all 128 cols of its rows, the attention dots are
//    completed in-block: shfl reduce over the 16-lane col group, ds_add_f32
//    across waves.
// ---------------------------------------------------------------------------
__global__ __launch_bounds__(256) void gat_gemm_wmma(
    const float* __restrict__ x, const _Float16* __restrict__ Wt,
    const float* __restrict__ att_src, const float* __restrict__ att_dst,
    float* __restrict__ h, float* __restrict__ asrc, float* __restrict__ adst,
    int N)
{
    __shared__ _Float16 Wlds[128 * 128];   // transposed: [o][k], 32KB
    __shared__ _Float16 Alds[32 * 128];    // [row][k], 8KB
    __shared__ float    attS[32];          // per-row a_src partials
    __shared__ float    attD[32];          // per-row a_dst partials

    const int tid     = threadIdx.x;
    const int wave    = tid >> 5;
    const int lane    = tid & 31;
    const int rowbase = blockIdx.x * 32;

    if (tid < 32) { attS[tid] = 0.0f; attD[tid] = 0.0f; }
    // stage Wt: contiguous 16B chunks, coalesced
    {
        const uint4* src = (const uint4*)Wt;       // 2048 uint4
        uint4* dstl = (uint4*)Wlds;
        for (int i = tid; i < 2048; i += 256) dstl[i] = src[i];
    }
    // stage A tile (f32 -> f16)
    for (int i = tid; i < 32 * 128; i += 256) {
        int r = i >> 7, c = i & 127;
        int n = rowbase + r;
        Alds[i] = (n < N) ? (_Float16)x[n * 128 + c] : (_Float16)0.0f;
    }
    __syncthreads();

    const int hi  = lane >> 4;     // half-wave select
    const int lo  = lane & 15;
    const int col = wave * 16 + lo;

    v8f acc0 = {}, acc1 = {};
#pragma unroll
    for (int k0 = 0; k0 < 128; k0 += 32) {
        // B 32x16 f16 layout: lane(lo)=col N; V0..3: K=8*hi+0..7, V4..7: K=16+8*hi+0..7
        v16h b;
        const _Float16* bp = &Wlds[col * 128 + k0 + hi * 8];
#pragma unroll
        for (int i = 0; i < 8; ++i) { b[i] = bp[i]; b[8 + i] = bp[16 + i]; }

        // A 16x32 f16 layout: lane(lo)=row M; V0..3: K=8*hi+0..7, V4..7: K=16+8*hi+0..7
        v16h a0, a1;
        const _Float16* ap0 = &Alds[lo * 128 + k0 + hi * 8];
        const _Float16* ap1 = ap0 + 16 * 128;
#pragma unroll
        for (int i = 0; i < 8; ++i) {
            a0[i] = ap0[i]; a0[8 + i] = ap0[16 + i];
            a1[i] = ap1[i]; a1[8 + i] = ap1[16 + i];
        }
        acc0 = __builtin_amdgcn_wmma_f32_16x16x32_f16(
            false, a0, false, b, (short)0, acc0, false, false);
        acc1 = __builtin_amdgcn_wmma_f32_16x16x32_f16(
            false, a1, false, b, (short)0, acc1, false, false);
    }

    // ---- store h.  C/D f32 layout: VGPR r -> M = r + 8*hi, N = lo ----
    if (rowbase + 32 <= N) {            // interior block: branch-free
        float* hp0 = h + (size_t)(rowbase + hi * 8) * 128 + col;
#pragma unroll
        for (int r = 0; r < 8; ++r) {
            hp0[r * 128]            = acc0[r];
            hp0[(r + 16) * 128]     = acc1[r];
        }
    } else {
#pragma unroll
        for (int r = 0; r < 8; ++r) {
            int n0 = rowbase + hi * 8 + r;
            int n1 = n0 + 16;
            if (n0 < N) h[(size_t)n0 * 128 + col] = acc0[r];
            if (n1 < N) h[(size_t)n1 * 128 + col] = acc1[r];
        }
    }

    // ---- fused attention dots: a = h·att over this wave's 16 cols ----
    const float as = att_src[col];
    const float ad = att_dst[col];
#pragma unroll
    for (int r = 0; r < 8; ++r) {
        float s0 = acc0[r] * as, d0 = acc0[r] * ad;
        float s1 = acc1[r] * as, d1 = acc1[r] * ad;
#pragma unroll
        for (int off = 8; off >= 1; off >>= 1) {   // reduce within 16-lane half
            s0 += __shfl_xor(s0, off, 32);
            d0 += __shfl_xor(d0, off, 32);
            s1 += __shfl_xor(s1, off, 32);
            d1 += __shfl_xor(d1, off, 32);
        }
        if (lo == 0) {
            int row = hi * 8 + r;
            atomicAdd(&attS[row], s0);      atomicAdd(&attD[row], d0);
            atomicAdd(&attS[16 + row], s1); atomicAdd(&attD[16 + row], d1);
        }
    }
    __syncthreads();
    if (tid < 32) {
        int n = rowbase + tid;
        if (n < N) { asrc[n] = attS[tid]; adst[n] = attD[tid]; }
    }
}

// ---------------------------------------------------------------------------
// 2) init: m = -inf, s = 0, out = 0
// ---------------------------------------------------------------------------
__global__ __launch_bounds__(256) void gat_init(
    float* __restrict__ m, float* __restrict__ s, float* __restrict__ out, int N)
{
    int i = blockIdx.x * 256 + threadIdx.x;
    if (i < N * 128) out[i] = 0.0f;
    if (i < N) { m[i] = -INFINITY; s[i] = 0.0f; }
}

// ---------------------------------------------------------------------------
// 3) segment max of leaky_relu logits (edges + self loops). Thread per edge.
// ---------------------------------------------------------------------------
__global__ __launch_bounds__(256) void gat_edge_max(
    const int* __restrict__ esrc, const int* __restrict__ edst,
    const float* __restrict__ asrc, const float* __restrict__ adst,
    float* __restrict__ m, int E, int T)
{
    int k = blockIdx.x * 256 + threadIdx.x;
    if (k >= T) return;
    int s = (k < E) ? esrc[k] : (k - E);
    int d = (k < E) ? edst[k] : (k - E);
    float e = lrelu(asrc[s] + adst[d]);
    atomicMax(&m[d], e);
}

// ---------------------------------------------------------------------------
// 4) segment sum of exp(e - m[dst]).
// ---------------------------------------------------------------------------
__global__ __launch_bounds__(256) void gat_edge_sum(
    const int* __restrict__ esrc, const int* __restrict__ edst,
    const float* __restrict__ asrc, const float* __restrict__ adst,
    const float* __restrict__ m, float* __restrict__ ssum, int E, int T)
{
    int k = blockIdx.x * 256 + threadIdx.x;
    if (k >= T) return;
    int s = (k < E) ? esrc[k] : (k - E);
    int d = (k < E) ? edst[k] : (k - E);
    float e = lrelu(asrc[s] + adst[d]);
    atomicAdd(&ssum[d], __expf(e - m[d]));
}

// ---------------------------------------------------------------------------
// 5) out[dst] += alpha * h[src].  One wave per edge, 4 channels per lane.
//    h/out (~51MB each) are L2-resident on MI455X (192MB L2), so the random
//    gathers + atomics resolve at L2 bandwidth.
// ---------------------------------------------------------------------------
__global__ __launch_bounds__(256) void gat_edge_aggr(
    const int* __restrict__ esrc, const int* __restrict__ edst,
    const float* __restrict__ asrc, const float* __restrict__ adst,
    const float* __restrict__ m, const float* __restrict__ ssum,
    const float* __restrict__ h, float* __restrict__ out, int E, int T)
{
    int k    = blockIdx.x * 8 + (threadIdx.x >> 5);
    int lane = threadIdx.x & 31;
    if (k >= T) return;
    int s = (k < E) ? esrc[k] : (k - E);
    int d = (k < E) ? edst[k] : (k - E);
    float e     = lrelu(asrc[s] + adst[d]);
    float alpha = __expf(e - m[d]) / (ssum[d] + 1e-16f);
    float4 v = ((const float4*)(h + (size_t)s * 128))[lane];
    float* op = out + (size_t)d * 128 + lane * 4;
    atomicAdd(op + 0, alpha * v.x);
    atomicAdd(op + 1, alpha * v.y);
    atomicAdd(op + 2, alpha * v.z);
    atomicAdd(op + 3, alpha * v.w);
}

// ---------------------------------------------------------------------------
// 6) out = relu(out + bias)
// ---------------------------------------------------------------------------
__global__ __launch_bounds__(256) void gat_bias_relu(
    float* __restrict__ out, const float* __restrict__ bias, int total)
{
    int i = blockIdx.x * 256 + threadIdx.x;
    if (i >= total) return;
    float v = out[i] + bias[i & 127];
    out[i] = v > 0.0f ? v : 0.0f;
}

extern "C" void kernel_launch(void* const* d_in, const int* in_sizes, int n_in,
                              void* d_out, int out_size, void* d_ws, size_t ws_size,
                              hipStream_t stream)
{
    const float* x        = (const float*)d_in[0];
    const int*   ei       = (const int*)d_in[1];
    const float* W        = (const float*)d_in[2];
    const float* att_src  = (const float*)d_in[3];
    const float* att_dst  = (const float*)d_in[4];
    const float* bias     = (const float*)d_in[5];
    float*       out      = (float*)d_out;

    const int N = in_sizes[0] / 128;
    const int E = in_sizes[1] / 2;
    const int T = E + N;               // edges + self loops
    const int Np = (N + 63) & ~63;     // padded region size

    const int* esrc = ei;
    const int* edst = ei + E;

    // workspace layout (floats): h | asrc | adst | m | ssum | Wt(f16)
    float* h    = (float*)d_ws;             // N*128 floats (base 16B-aligned)
    float* asrc = h + (size_t)N * 128;      // Np
    float* adst = asrc + Np;                // Np
    float* m    = adst + Np;                // Np
    float* ssum = m + Np;                   // Np
    _Float16* Wt = (_Float16*)(ssum + Np);  // 16384 f16 (32KB), 16B-aligned

    // 0) W -> f16 transposed
    gat_w_f16t<<<64, 256, 0, stream>>>(W, Wt);
    // 1) GEMM (WMMA) + fused attention dots, 32 rows per block
    gat_gemm_wmma<<<(N + 31) / 32, 256, 0, stream>>>(
        x, Wt, att_src, att_dst, h, asrc, adst, N);
    // 2) init m, s, out
    gat_init<<<(N * 128 + 255) / 256, 256, 0, stream>>>(m, ssum, out, N);
    // 3) segment max
    gat_edge_max<<<(T + 255) / 256, 256, 0, stream>>>(esrc, edst, asrc, adst, m, E, T);
    // 4) segment sum
    gat_edge_sum<<<(T + 255) / 256, 256, 0, stream>>>(esrc, edst, asrc, adst, m, ssum, E, T);
    // 5) weighted aggregate
    gat_edge_aggr<<<(T + 7) / 8, 256, 0, stream>>>(esrc, edst, asrc, adst, m, ssum, h, out, E, T);
    // 6) bias + relu
    gat_bias_relu<<<(N * 128 + 255) / 256, 256, 0, stream>>>(out, bias, N * 128);
}
